// LlamaDecoderLayer_70918499991962
// MI455X (gfx1250) — matrix operand order repaired and checked by reading, not compiled, not run on hardware
//
#include <hip/hip_runtime.h>
#include <hip/hip_bf16.h>

// ---------------------------------------------------------------------------
// Llama decoder layer on MI455X (gfx1250, wave32, WMMA).
// bf16 data path with fp32 WMMA accumulation; flash attention (no S x S in HBM).
// CDNA5 data movement: TDM tensor_load_to_lds for GEMM A-tiles,
// global_load_async_to_lds_b128 for flash K-tiles.
// ---------------------------------------------------------------------------

#define HID   2048
#define NHEAD 16
#define HD    128
#define SEQ   2048
#define BATCH 2
#define MROWS (BATCH * SEQ)   // 4096

typedef __bf16 bf16_t;
typedef __bf16 bf16x16 __attribute__((ext_vector_type(16)));
typedef float  f32x8   __attribute__((ext_vector_type(8)));
typedef unsigned int uint32x4 __attribute__((ext_vector_type(4)));
typedef unsigned int uint32x8 __attribute__((ext_vector_type(8)));

union FragB { uint4 u[2]; bf16x16 v; };

__device__ __forceinline__ f32x8 wmma_bf16(bf16x16 a, bf16x16 b, f32x8 c) {
    // D = A(16x32 bf16) * B(32x16 bf16) + C(16x16 f32)
    return __builtin_amdgcn_wmma_f32_16x16x32_bf16(
        /*neg_a=*/false, a, /*neg_b=*/false, b,
        /*c_mod=*/(short)0, c, /*reuse_a=*/false, /*reuse_b=*/false);
}

// --- CDNA5 async / TDM helpers (inline asm per CDNA5 ISA) ---

// Per-lane async copy of 16 bytes global -> LDS, tracked on ASYNCcnt.
__device__ __forceinline__ void async_load_b128(unsigned int lds_off,
                                                unsigned long long gaddr) {
    asm volatile("global_load_async_to_lds_b128 %0, %1, off"
                 :: "v"(lds_off), "v"(gaddr) : "memory");
}
__device__ __forceinline__ void wait_asynccnt0() {
    asm volatile("s_wait_asynccnt 0x0" ::: "memory");
}

// TDM: DMA a 2D tile (tile_d0 x tile_d1 elements of 2 bytes) from a row-major
// tensor (row length = stride elements) into LDS at lds_off. 2-D => groups 2/3 NULL.
__device__ __forceinline__ void tdm_load_tile_2d_bf16(unsigned int lds_off,
                                                      unsigned long long gaddr,
                                                      unsigned int tensor_d0,
                                                      unsigned int tensor_d1,
                                                      unsigned int tile_d0,
                                                      unsigned int tile_d1,
                                                      unsigned int stride_d0) {
    uint32x4 g0;
    g0.x = 1u;                                             // count=1, user desc
    g0.y = lds_off;                                        // lds_addr (bytes)
    g0.z = (unsigned int)gaddr;                            // global_addr[31:0]
    g0.w = ((unsigned int)(gaddr >> 32) & 0x01FFFFFFu)     // global_addr[56:32]
           | (2u << 30);                                   // type=2 ("image")
    uint32x8 g1;
    g1.s0 = (1u << 16);                                    // data_size=1 (2 bytes)
    g1.s1 = (tensor_d0 & 0xFFFFu) << 16;                   // tensor_dim0[15:0]
    g1.s2 = (tensor_d0 >> 16) | ((tensor_d1 & 0xFFFFu) << 16); // dim0 hi / dim1 lo
    g1.s3 = (tensor_d1 >> 16) | (tile_d0 << 16);           // dim1 hi / tile_dim0
    g1.s4 = tile_d1;                                       // tile_dim1 (tile_dim2=0)
    g1.s5 = stride_d0;                                     // tensor_dim0_stride[31:0]
    g1.s6 = 0u;                                            // stride hi / dim1_stride lo
    g1.s7 = 0u;
    asm volatile("tensor_load_to_lds %0, %1, null, null"
                 :: "s"(g0), "s"(g1) : "memory");
}
__device__ __forceinline__ void wait_tensorcnt0() {
#if defined(__has_builtin)
#if __has_builtin(__builtin_amdgcn_s_wait_tensorcnt)
    __builtin_amdgcn_s_wait_tensorcnt(0);
#define TENSORCNT_DONE 1
#endif
#endif
#ifndef TENSORCNT_DONE
    asm volatile("s_wait_tensorcnt 0x0" ::: "memory");
#endif
}

__device__ __forceinline__ unsigned int lds_offset_of(const void* p) {
    // Generic pointers into the LDS aperture carry the LDS byte offset in the
    // low 32 bits (flat aperture mapping, CDNA5 ISA).
    return (unsigned int)(unsigned long long)p;
}

// ---------------------------------------------------------------------------
// fp32 -> bf16 conversion (weights)
// ---------------------------------------------------------------------------
__global__ __launch_bounds__(256)
void cvt_bf16_kernel(const float* __restrict__ in, bf16_t* __restrict__ out, int n) {
    int i = (blockIdx.x * 256 + threadIdx.x) * 4;
    if (i + 3 < n) {
        float4 f = *(const float4*)(in + i);
        out[i + 0] = (bf16_t)f.x;
        out[i + 1] = (bf16_t)f.y;
        out[i + 2] = (bf16_t)f.z;
        out[i + 3] = (bf16_t)f.w;
    }
}

// ---------------------------------------------------------------------------
// RMSNorm: one block per row (2048 f32), bf16 output
// ---------------------------------------------------------------------------
__global__ __launch_bounds__(256)
void rmsnorm_kernel(const float* __restrict__ X, const float* __restrict__ W,
                    bf16_t* __restrict__ Y) {
    __shared__ float red[8];
    const int row = blockIdx.x, tid = threadIdx.x;
    const float* x = X + (size_t)row * HID;
    float v[8];
    float ss = 0.f;
#pragma unroll
    for (int i = 0; i < 8; ++i) { v[i] = x[tid + i * 256]; ss += v[i] * v[i]; }
#pragma unroll
    for (int off = 16; off > 0; off >>= 1) ss += __shfl_xor(ss, off, 32);
    if ((tid & 31) == 0) red[tid >> 5] = ss;
    __syncthreads();
    float tot = 0.f;
#pragma unroll
    for (int i = 0; i < 8; ++i) tot += red[i];
    const float inv = rsqrtf(tot * (1.0f / HID) + 1e-6f);
#pragma unroll
    for (int i = 0; i < 8; ++i)
        Y[(size_t)row * HID + tid + i * 256] = (bf16_t)(v[i] * inv * W[tid + i * 256]);
}

// ---------------------------------------------------------------------------
// WMMA GEMM: C[M,N] = A[M,K](bf16,row-major) @ B[K,N](bf16,row-major)
// Block tile 128x128, 8 waves -> 32x64 each (2x4 wmma tiles), K-step 32.
// A tile staged by TDM (tensor_load_to_lds, one desc per K-step, wave 0).
// EPI: 0 = store bf16, 1 = store f32, 2 = store f32 + residual R
// ---------------------------------------------------------------------------
template <int EPI>
__global__ __launch_bounds__(256)
void gemm_kernel(const bf16_t* __restrict__ A, const bf16_t* __restrict__ B,
                 void* __restrict__ C, const float* __restrict__ R,
                 int M, int N, int K) {
    __shared__ __align__(16) bf16_t As[128][32];   // A tile, row-major (TDM dest)
    __shared__ __align__(16) bf16_t Bt[128][32];   // B tile, transposed: Bt[n][k]
    const int tid = threadIdx.x, wave = tid >> 5, lane = tid & 31;
    const int bm = blockIdx.x * 128, bn = blockIdx.y * 128;
    const int wm = (wave & 3) * 32, wn = (wave >> 2) * 64;

    const f32x8 z = {0.f, 0.f, 0.f, 0.f, 0.f, 0.f, 0.f, 0.f};
    f32x8 acc[2][4];
#pragma unroll
    for (int mt = 0; mt < 2; ++mt)
#pragma unroll
        for (int nt = 0; nt < 4; ++nt) acc[mt][nt] = z;

    const int brow = tid >> 3, bseg = (tid & 7) * 16;   // B: 8 threads/row
    const int fm  = lane & 15;                          // fragment M (A) / N (B)
    const int fka = (lane >> 4) * 8;                    // A-frag K base (ISA layout)
    const int fkb = (lane >> 4) * 16;                   // unused (B frags load with fka)
    const unsigned int lds_As = lds_offset_of(&As[0][0]);

    for (int k0 = 0; k0 < K; k0 += 32) {
        // --- TDM: A tile rows [bm, bm+128), cols [k0, k0+32) -> As (row-major)
        if (wave == 0) {
            unsigned long long ga =
                (unsigned long long)(A + (size_t)bm * K + k0);
            tdm_load_tile_2d_bf16(lds_As, ga, (unsigned)K, (unsigned)M,
                                  32u, 128u, (unsigned)K);
            wait_tensorcnt0();
        }
        // --- B tile: global -> VGPR -> transposed LDS stores
        const bf16_t* bsrc = B + (size_t)(k0 + brow) * N + bn + bseg;
        uint4 b0 = ((const uint4*)bsrc)[0];
        uint4 b1 = ((const uint4*)bsrc)[1];
        if (k0 + 32 < K)  // prefetch next K-tile into L2 (global_prefetch_b8)
            __builtin_prefetch(B + (size_t)(k0 + 32 + brow) * N + bn + bseg, 0, 3);
        {
            bf16_t tmp[16];
            *(uint4*)&tmp[0] = b0;
            *(uint4*)&tmp[8] = b1;
#pragma unroll
            for (int i = 0; i < 16; ++i) Bt[bseg + i][brow] = tmp[i];
        }
        __syncthreads();

        bf16x16 af[2], bfr[4];
#pragma unroll
        for (int mt = 0; mt < 2; ++mt) {
            FragB f;
            f.u[0] = *(const uint4*)&As[wm + mt * 16 + fm][fka];
            f.u[1] = *(const uint4*)&As[wm + mt * 16 + fm][fka + 16];
            af[mt] = f.v;
        }
#pragma unroll
        for (int nt = 0; nt < 4; ++nt) {
            FragB f;
            f.u[0] = *(const uint4*)&Bt[wn + nt * 16 + fm][fka];
            f.u[1] = *(const uint4*)&Bt[wn + nt * 16 + fm][fka + 16];
            bfr[nt] = f.v;
        }
#pragma unroll
        for (int mt = 0; mt < 2; ++mt)
#pragma unroll
            for (int nt = 0; nt < 4; ++nt)
                acc[mt][nt] = wmma_bf16(af[mt], bfr[nt], acc[mt][nt]);
        __syncthreads();
    }

    const int cn = lane & 15, cm = (lane >> 4) * 8;
#pragma unroll
    for (int mt = 0; mt < 2; ++mt)
#pragma unroll
        for (int nt = 0; nt < 4; ++nt)
#pragma unroll
            for (int r = 0; r < 8; ++r) {
                const size_t idx =
                    (size_t)(bm + wm + mt * 16 + cm + r) * N + (bn + wn + nt * 16 + cn);
                const float val = acc[mt][nt][r];
                if (EPI == 0)      ((bf16_t*)C)[idx] = (bf16_t)val;
                else if (EPI == 1) ((float*)C)[idx]  = val;
                else               ((float*)C)[idx]  = val + R[idx];
            }
}

// ---------------------------------------------------------------------------
// RoPE on Q and K in-place ([B,S,NH,HD] bf16)
// ---------------------------------------------------------------------------
__global__ __launch_bounds__(256)
void rope_kernel(bf16_t* __restrict__ Q, bf16_t* __restrict__ Kp, size_t npairs) {
    const size_t idx = (size_t)blockIdx.x * 256 + threadIdx.x;
    if (idx >= npairs) return;
    const int i = (int)(idx & 63);                 // rotary pair index
    const size_t rem = idx >> 6;                   // (b*S + s)*NH + h
    const int s = (int)((rem / NHEAD) % SEQ);
    const size_t off = rem * HD;
    const float invf = __expf(-(float)i * (9.210340371976184f / 64.0f)); // 10000^(-i/64)
    float sn, c;
    __sincosf((float)s * invf, &sn, &c);
    {
        float q1 = (float)Q[off + i], q2 = (float)Q[off + i + 64];
        Q[off + i]      = (bf16_t)(q1 * c - q2 * sn);
        Q[off + i + 64] = (bf16_t)(q2 * c + q1 * sn);
    }
    {
        float k1 = (float)Kp[off + i], k2 = (float)Kp[off + i + 64];
        Kp[off + i]      = (bf16_t)(k1 * c - k2 * sn);
        Kp[off + i + 64] = (bf16_t)(k2 * c + k1 * sn);
    }
}

// ---------------------------------------------------------------------------
// Flash attention: grid (S/128, NH, B); 8 waves, 16 q-rows/wave; 64-key blocks.
// K tile staged with per-lane async global->LDS copies (ASYNCcnt path).
// ---------------------------------------------------------------------------
__global__ __launch_bounds__(256)
void flash_attn_kernel(const bf16_t* __restrict__ Q, const bf16_t* __restrict__ K,
                       const bf16_t* __restrict__ V, bf16_t* __restrict__ O) {
    __shared__ __align__(16) bf16_t Ks[64][128];      // keys, row-major over d
    __shared__ __align__(16) bf16_t Vt[128][64];      // V transposed: Vt[d][k]
    __shared__ __align__(16) bf16_t Ps[8][16][64];    // per-wave P staging
    const int tid = threadIdx.x, wave = tid >> 5, lane = tid & 31;
    const int h = blockIdx.y, b = blockIdx.z;
    const int q0 = blockIdx.x * 128 + wave * 16;
    const size_t base = ((size_t)b * SEQ) * HID + (size_t)h * HD;
    const int fm = lane & 15, fka = (lane >> 4) * 8, fkb = (lane >> 4) * 16;
    const int cn = lane & 15, cm = (lane >> 4) * 8;

    bf16x16 qf[4];
#pragma unroll
    for (int d = 0; d < 4; ++d) {
        const bf16_t* qp = Q + base + (size_t)(q0 + fm) * HID + d * 32;
        FragB f;
        f.u[0] = *(const uint4*)(qp + fka);
        f.u[1] = *(const uint4*)(qp + fka + 16);
        qf[d] = f.v;
    }

    const f32x8 z = {0.f, 0.f, 0.f, 0.f, 0.f, 0.f, 0.f, 0.f};
    f32x8 ctx[8];
#pragma unroll
    for (int t = 0; t < 8; ++t) ctx[t] = z;
    float m_i[8], l_i[8];
#pragma unroll
    for (int r = 0; r < 8; ++r) { m_i[r] = -1e30f; l_i[r] = 0.f; }
    const float scale = 0.08838834764831845f;  // 1/sqrt(128)
    const int krow = tid >> 2, dseg = (tid & 3) * 32;

    for (int j0 = 0; j0 < SEQ; j0 += 64) {
        {   // stage K block with async global->LDS (row-major, 4 x 16B per lane)
            const bf16_t* src = K + base + (size_t)(j0 + krow) * HID + dseg;
#pragma unroll
            for (int j = 0; j < 4; ++j)
                async_load_b128(lds_offset_of(&Ks[krow][dseg + 8 * j]),
                                (unsigned long long)(src + 8 * j));
        }
        {   // stage V transposed (PV B-fragment wants keys contiguous per d)
            const bf16_t* src = V + base + (size_t)(j0 + krow) * HID + dseg;
            uint4 d0 = ((const uint4*)src)[0], d1 = ((const uint4*)src)[1];
            uint4 d2 = ((const uint4*)src)[2], d3 = ((const uint4*)src)[3];
            bf16_t tmp[32];
            *(uint4*)&tmp[0] = d0; *(uint4*)&tmp[8] = d1;
            *(uint4*)&tmp[16] = d2; *(uint4*)&tmp[24] = d3;
#pragma unroll
            for (int i = 0; i < 32; ++i) Vt[dseg + i][krow] = tmp[i];
        }
        wait_asynccnt0();
        __syncthreads();

        // scores: 4 key tiles x 4 d-steps of WMMA
        f32x8 sc[4];
#pragma unroll
        for (int kt = 0; kt < 4; ++kt) {
            sc[kt] = z;
#pragma unroll
            for (int d = 0; d < 4; ++d) {
                FragB f;
                const bf16_t* kp = &Ks[kt * 16 + fm][d * 32 + fka];
                f.u[0] = *(const uint4*)kp;
                f.u[1] = *(const uint4*)(kp + 16);
                sc[kt] = wmma_bf16(qf[d], f.v, sc[kt]);
            }
        }

        // online softmax (row reductions stay inside 16-lane halves: xor 1..8)
        float pv[4][8];
#pragma unroll
        for (int r = 0; r < 8; ++r) {
            float mx = -1e30f;
#pragma unroll
            for (int kt = 0; kt < 4; ++kt) {
                float s = sc[kt][r] * scale;
                pv[kt][r] = s;
                mx = fmaxf(mx, s);
            }
#pragma unroll
            for (int off = 8; off > 0; off >>= 1) mx = fmaxf(mx, __shfl_xor(mx, off, 32));
            const float mn = fmaxf(m_i[r], mx);
            const float corr = __expf(m_i[r] - mn);
            m_i[r] = mn;
            float rs = 0.f;
#pragma unroll
            for (int kt = 0; kt < 4; ++kt) {
                float e = __expf(pv[kt][r] - mn);
                pv[kt][r] = e;
                rs += e;
            }
#pragma unroll
            for (int off = 8; off > 0; off >>= 1) rs += __shfl_xor(rs, off, 32);
            l_i[r] = l_i[r] * corr + rs;
#pragma unroll
            for (int t = 0; t < 8; ++t) ctx[t][r] *= corr;
        }

        // C-layout P -> LDS -> A-fragment re-layout (wave-private, in-order DS)
#pragma unroll
        for (int kt = 0; kt < 4; ++kt)
#pragma unroll
            for (int r = 0; r < 8; ++r)
                Ps[wave][cm + r][kt * 16 + cn] = (bf16_t)pv[kt][r];

        bf16x16 pf[2];
#pragma unroll
        for (int ks = 0; ks < 2; ++ks) {
            FragB f;
            const bf16_t* pp = &Ps[wave][fm][ks * 32 + fka];
            f.u[0] = *(const uint4*)pp;
            f.u[1] = *(const uint4*)(pp + 16);
            pf[ks] = f.v;
        }
#pragma unroll
        for (int t = 0; t < 8; ++t)
#pragma unroll
            for (int ks = 0; ks < 2; ++ks) {
                FragB f;
                const bf16_t* vp = &Vt[t * 16 + fm][ks * 32 + fka];
                f.u[0] = *(const uint4*)vp;
                f.u[1] = *(const uint4*)(vp + 16);
                ctx[t] = wmma_bf16(pf[ks], f.v, ctx[t]);
            }
        __syncthreads();
    }

#pragma unroll
    for (int t = 0; t < 8; ++t)
#pragma unroll
        for (int r = 0; r < 8; ++r) {
            const float o = ctx[t][r] / l_i[r];
            O[base + (size_t)(q0 + cm + r) * HID + t * 16 + cn] = (bf16_t)o;
        }
}

// ---------------------------------------------------------------------------
// SwiGLU activation: y = silu(g) * u, bf16 out
// ---------------------------------------------------------------------------
__global__ __launch_bounds__(256)
void silu_mul_kernel(const float* __restrict__ G, const float* __restrict__ U,
                     bf16_t* __restrict__ Y, int n) {
    int i = (blockIdx.x * 256 + threadIdx.x) * 4;
    if (i + 3 < n) {
#pragma unroll
        for (int j = 0; j < 4; ++j) {
            float g = G[i + j];
            float s = g / (1.f + __expf(-g));
            Y[i + j] = (bf16_t)(s * U[i + j]);
        }
    }
}

// ---------------------------------------------------------------------------
// Orchestration
// ---------------------------------------------------------------------------
extern "C" void kernel_launch(void* const* d_in, const int* in_sizes, int n_in,
                              void* d_out, int out_size, void* d_ws, size_t ws_size,
                              hipStream_t stream) {
    (void)in_sizes; (void)n_in; (void)out_size; (void)ws_size;
    const float* hidden = (const float*)d_in[0];
    const float* W_f32[7] = {
        (const float*)d_in[1], (const float*)d_in[2], (const float*)d_in[3],
        (const float*)d_in[4], (const float*)d_in[5], (const float*)d_in[6],
        (const float*)d_in[7]
    };  // Wq, Wk, Wv, Wo, w1, w2, w3
    const float* ln1 = (const float*)d_in[8];
    const float* ln2 = (const float*)d_in[9];

    char* ws = (char*)d_ws;
    size_t off = 0;
    auto alloc = [&](size_t bytes) {
        char* p = ws + off;
        off += (bytes + 255) & ~(size_t)255;
        return (void*)p;
    };
    const size_t WB  = (size_t)HID * HID * sizeof(bf16_t);    // 8 MB
    const size_t ABF = (size_t)MROWS * HID * sizeof(bf16_t);  // 16 MB
    const size_t AF  = (size_t)MROWS * HID * sizeof(float);   // 32 MB

    bf16_t* Wb[7];
    for (int i = 0; i < 7; ++i) Wb[i] = (bf16_t*)alloc(WB);
    bf16_t* xn   = (bf16_t*)alloc(ABF);
    bf16_t* qb   = (bf16_t*)alloc(ABF);
    bf16_t* kb   = (bf16_t*)alloc(ABF);
    bf16_t* vb   = (bf16_t*)alloc(ABF);
    bf16_t* ctxb = (bf16_t*)alloc(ABF);
    bf16_t* yb   = (bf16_t*)alloc(ABF);
    bf16_t* actb = (bf16_t*)alloc(ABF);
    float*  hbuf = (float*)alloc(AF);
    float*  gate = (float*)alloc(AF);
    float*  up   = (float*)alloc(AF);

    const int WELEM = HID * HID;
    for (int i = 0; i < 7; ++i)
        cvt_bf16_kernel<<<WELEM / 1024, 256, 0, stream>>>(W_f32[i], Wb[i], WELEM);

    // attention block
    rmsnorm_kernel<<<MROWS, 256, 0, stream>>>(hidden, ln1, xn);
    dim3 gg(MROWS / 128, HID / 128);
    gemm_kernel<0><<<gg, 256, 0, stream>>>(xn, Wb[0], qb, nullptr, MROWS, HID, HID);
    gemm_kernel<0><<<gg, 256, 0, stream>>>(xn, Wb[1], kb, nullptr, MROWS, HID, HID);
    gemm_kernel<0><<<gg, 256, 0, stream>>>(xn, Wb[2], vb, nullptr, MROWS, HID, HID);

    const size_t npairs = (size_t)MROWS * NHEAD * (HD / 2);   // 4,194,304
    rope_kernel<<<(unsigned)(npairs / 256), 256, 0, stream>>>(qb, kb, npairs);

    flash_attn_kernel<<<dim3(SEQ / 128, NHEAD, BATCH), 256, 0, stream>>>(qb, kb, vb, ctxb);

    // attn_out @ Wo + residual -> h
    gemm_kernel<2><<<gg, 256, 0, stream>>>(ctxb, Wb[3], hbuf, hidden, MROWS, HID, HID);

    // MLP block
    rmsnorm_kernel<<<MROWS, 256, 0, stream>>>(hbuf, ln2, yb);
    gemm_kernel<1><<<gg, 256, 0, stream>>>(yb, Wb[4], gate, nullptr, MROWS, HID, HID);  // w1
    gemm_kernel<1><<<gg, 256, 0, stream>>>(yb, Wb[6], up,   nullptr, MROWS, HID, HID);  // w3
    const int NACT = MROWS * HID;
    silu_mul_kernel<<<NACT / 1024, 256, 0, stream>>>(gate, up, actb, NACT);
    gemm_kernel<2><<<gg, 256, 0, stream>>>(actb, Wb[5], (float*)d_out, hbuf,
                                           MROWS, HID, HID);  // w2 + residual
}
